// FCCaps_8546984919199
// MI455X (gfx1250) — compile-verified, matrix operand order checked by reading, not run on hardware
//
#include <hip/hip_runtime.h>
#include <stdint.h>

// Problem constants (match reference)
#define BB 32      // batch
#define NI 2048    // input capsules
#define KO 128     // output capsules
#define CI 16      // in caps dim
#define CO 16      // out caps dim

typedef __attribute__((ext_vector_type(16))) __bf16 v16bf;
typedef __attribute__((ext_vector_type(8)))  float  v8f;

union BFrag { v16bf v; uint32_t u[8]; uint4 q[2]; };

static __device__ __forceinline__ uint32_t pack2(float a, float b) {
  const uint16_t lo = __builtin_bit_cast(uint16_t, (__bf16)a);
  const uint16_t hi = __builtin_bit_cast(uint16_t, (__bf16)b);
  return ((uint32_t)hi << 16) | (uint32_t)lo;
}

__global__ void zero_kernel(float* __restrict__ p, int n) {
  int i = blockIdx.x * blockDim.x + threadIdx.x;
  if (i < n) p[i] = 0.0f;
}

// ---------------------------------------------------------------------------
// One-shot W swizzle: fp32 W[i,k,d,c] -> bf16, fragment-native layout.
// Chunk per (p = i/2, k): 32 lanes x 32B; lane l (h=l>>4, n=l&15) holds
//   [ W[2p,k,n, 8h..8h+7] | W[2p+1,k,n, 8h..8h+7] ]   (bf16 pairs, j order)
// which is exactly that lane's v16bf B operand for the 16x16x32 WMMA with
// flattened K=(i-pair, c). Layout is k-major then p so v-pass streams p.
// ---------------------------------------------------------------------------
__global__ __launch_bounds__(256) void swizzle_kernel(
    const float* __restrict__ W, uint32_t* __restrict__ Wswz) {
  const size_t t = (size_t)blockIdx.x * 256 + threadIdx.x;  // 4,194,304 chunks
  const int lane = (int)(t & 31);
  const int p    = (int)((t >> 5) & 1023);
  const int k    = (int)(t >> 15);
  const int h = lane >> 4, n = lane & 15;
  const float* s0 = W + (((size_t)(2 * p)     * KO + k) * CO + n) * CI + 8 * h;
  const float* s1 = W + (((size_t)(2 * p + 1) * KO + k) * CO + n) * CI + 8 * h;
  uint4 q0, q1;
  q0.x = pack2(s0[0], s0[1]); q0.y = pack2(s0[2], s0[3]);
  q0.z = pack2(s0[4], s0[5]); q0.w = pack2(s0[6], s0[7]);
  q1.x = pack2(s1[0], s1[1]); q1.y = pack2(s1[2], s1[3]);
  q1.z = pack2(s1[4], s1[5]); q1.w = pack2(s1[6], s1[7]);
  uint4* d = (uint4*)Wswz + (((size_t)k * 1024 + p) * 32 + lane) * 2;
  d[0] = q0; d[1] = q1;
}

// ---------------------------------------------------------------------------
// v-pass: vraw[b,k,d] = sum_i c[b,i,k] * sum_c x[b,i,c] * W[i,k,d,c]
// GEMM over flattened K=(i,c); each bf16 16x16x32 WMMA eats 2 i's, full K.
// grid = (KO, 2 b-tiles); block = 256 (8 waves) exclusively owns (k, b-tile).
// ---------------------------------------------------------------------------
template <bool SWZ>
__global__ __launch_bounds__(256) void vpass_kernel(
    const float* __restrict__ x, const float* __restrict__ W,
    const uint32_t* __restrict__ Wswz,
    const float* __restrict__ cbuf, float cuni, float* __restrict__ vraw) {
  const int k     = blockIdx.x;
  const int btile = blockIdx.y;
  const int tid = threadIdx.x, w = tid >> 5, lane = tid & 31;
  const int h = lane >> 4, m = lane & 15;   // A row (b) / B col (d)
  const int b = btile * 16 + m;

  v8f acc;
  #pragma unroll
  for (int r = 0; r < 8; ++r) acc[r] = 0.0f;

  for (int s = w; s < NI / 2; s += 8) {     // step = one i-pair (K = 32)
    const int i0 = 2 * s;
    float cv0, cv1;
    if (cbuf) {
      cv0 = cbuf[((size_t)b * NI + i0) * KO + k];
      cv1 = cbuf[((size_t)b * NI + i0 + 1) * KO + k];
    } else { cv0 = cuni; cv1 = cuni; }

    BFrag fa, fb;
    const float* xp0 = x + ((size_t)b * NI + i0) * CI + 8 * h;
    #pragma unroll
    for (int e = 0; e < 8; ++e) {           // A': c-scaled x, native bf16 cvt
      fa.v[e]     = (__bf16)(xp0[e] * cv0);
      fa.v[8 + e] = (__bf16)(xp0[CI + e] * cv1);
    }
    if (SWZ) {
      const uint4* wp = (const uint4*)Wswz + (((size_t)k * 1024 + s) * 32 + lane) * 2;
      fb.q[0] = wp[0];                      // fragment-native: zero repack VALU
      fb.q[1] = wp[1];
      if (s + 8 < NI / 2)
        __builtin_prefetch((const uint4*)Wswz +
                           (((size_t)k * 1024 + s + 8) * 32 + lane) * 2, 0, 1);
    } else {
      const float* wp0 = W + (((size_t)i0 * KO + k) * CO + m) * CI + 8 * h;
      #pragma unroll
      for (int e = 0; e < 8; ++e) {
        fb.v[e]     = (__bf16)wp0[e];
        fb.v[8 + e] = (__bf16)wp0[(size_t)KO * CO * CI + e];
      }
    }
    acc = __builtin_amdgcn_wmma_f32_16x16x32_bf16(
        false, fa.v, false, fb.v, (short)0, acc, false, false);
  }

  // cross-wave reduce (C layout: M = r + 8*half, N = lane&15)
  __shared__ float red[8 * 256];
  #pragma unroll
  for (int r = 0; r < 8; ++r) red[w * 256 + r * 32 + lane] = acc[r];
  __syncthreads();
  {
    float sum = 0.0f;
    #pragma unroll
    for (int ww = 0; ww < 8; ++ww) sum += red[ww * 256 + tid];
    const int r2 = tid >> 5, l2 = tid & 31;
    const int M = r2 + 8 * (l2 >> 4);
    const int d = l2 & 15;
    vraw[((size_t)(btile * 16 + M) * KO + k) * CO + d] = sum;
  }
}

// ---------------------------------------------------------------------------
// softmax over k per (b,i) row + accumulate S[b,k] = sum_i c
// ---------------------------------------------------------------------------
__global__ __launch_bounds__(128) void softmax_kernel(
    const float* __restrict__ blog, float* __restrict__ cbuf,
    float* __restrict__ S) {
  const int row = blockIdx.x;          // b*NI + i
  const int b   = row / NI;
  const int k   = threadIdx.x;
  __shared__ float sd[128];
  const size_t base = (size_t)row * KO;
  const float v = blog[base + k];
  sd[k] = v; __syncthreads();
  for (int off = 64; off > 0; off >>= 1) {
    if (k < off) sd[k] = fmaxf(sd[k], sd[k + off]);
    __syncthreads();
  }
  const float mx = sd[0]; __syncthreads();
  const float e = __expf(v - mx);
  sd[k] = e; __syncthreads();
  for (int off = 64; off > 0; off >>= 1) {
    if (k < off) sd[k] += sd[k + off];
    __syncthreads();
  }
  const float c = e / sd[0];
  cbuf[base + k] = c;
  atomicAdd(&S[(size_t)b * KO + k], c);
}

// ---------------------------------------------------------------------------
// finalize: v = squash(vraw / S); optionally write poses+activations
// ---------------------------------------------------------------------------
__global__ __launch_bounds__(256) void finalize_kernel(
    const float* __restrict__ vraw, const float* __restrict__ S,
    float* __restrict__ vbuf, float* __restrict__ out) {
  const int idx = blockIdx.x * blockDim.x + threadIdx.x;  // (b,k)
  if (idx >= BB * KO) return;
  const float denom = S ? S[idx] : 1.0f;
  float vp[16]; float s2 = 0.0f;
  #pragma unroll
  for (int d = 0; d < 16; ++d) {
    const float t = vraw[(size_t)idx * 16 + d] / denom;
    vp[d] = t; s2 += t * t;
  }
  const float scale = sqrtf(s2) / (0.5f + s2);
  #pragma unroll
  for (int d = 0; d < 16; ++d) {
    const float vo = scale * vp[d];
    vbuf[(size_t)idx * 16 + d] = vo;
    if (out) out[(size_t)idx * 16 + d] = vo;                    // poses
  }
  if (out) out[(size_t)BB * KO * CO + idx] = s2 / (0.5f + s2);  // activations
}

// ---------------------------------------------------------------------------
// dd-pass: recompute u_hat tile per (b-tile, i, k) via WMMA (K = c padded),
// reduce s2=|u|^2 and t=u.v per row through per-wave LDS scratch (C rows are
// d-contiguous in scratch), then blog += 1 - |u_sq - v|^2.
// grid = (2 b-tiles, 4 k-groups of 32, 32 i-chunks of 64); block 256 (8 waves)
// LDS: v slice 32KB + |v|^2 2KB + wave scratch 8KB = 42KB.
// ---------------------------------------------------------------------------
template <bool SWZ>
__global__ __launch_bounds__(256) void ddpass_kernel(
    const float* __restrict__ x, const float* __restrict__ W,
    const uint32_t* __restrict__ Wswz,
    const float* __restrict__ vbuf, float* __restrict__ blog) {
  const int btile = blockIdx.x;
  const int kg    = blockIdx.y;
  const int ich   = blockIdx.z;
  const int tid = threadIdx.x, w = tid >> 5, lane = tid & 31;
  const int h = lane >> 4, m = lane & 15;

  __shared__ float vs[16 * 32 * 16];     // v[b_local][k_local][d]
  __shared__ float vsq[16 * 32];         // |v|^2
  __shared__ float sc[8 * 256];          // per-wave C-tile scratch
  for (int q = tid; q < 16 * 32 * 16; q += 256) {
    const int mb = q >> 9, rem = q & 511, kk = rem >> 4, d = rem & 15;
    vs[q] = vbuf[(((size_t)(btile * 16 + mb)) * KO + (kg * 32 + kk)) * CO + d];
  }
  __syncthreads();
  for (int q = tid; q < 16 * 32; q += 256) {
    float s = 0.0f;
    #pragma unroll
    for (int d = 0; d < 16; ++d) { const float t = vs[q * 16 + d]; s += t * t; }
    vsq[q] = s;
  }
  __syncthreads();

  float* mysc = sc + w * 256;
  const int b  = btile * 16 + m;
  const int br = lane & 15;              // row this lane reduces
  const int hh = lane >> 4;              // which 8 d's
  const float* crow0 = mysc + (br & 7) * 32 + (br >> 3) * 16 + hh * 8;

  for (int ii = w; ii < 64; ii += 8) {
    const int i = ich * 64 + ii;
    BFrag fa;                            // A = x[b,i,c], K padded 16->32
    const float* xp = x + ((size_t)b * NI + i) * CI + 8 * h;
    #pragma unroll
    for (int e = 0; e < 8; ++e) fa.v[e] = (__bf16)xp[e];
    fa.u[4] = 0; fa.u[5] = 0; fa.u[6] = 0; fa.u[7] = 0;

    for (int kk = 0; kk < 32; ++kk) {
      const int k = kg * 32 + kk;
      BFrag fb;
      if (SWZ) {
        const uint4* wp = (const uint4*)Wswz +
            (((size_t)k * 1024 + (i >> 1)) * 32 + lane) * 2 + (i & 1);
        fb.q[0] = *wp;                   // one b128, zero conversion VALU
      } else {
        const float* wp0 = W + (((size_t)i * KO + k) * CO + m) * CI + 8 * h;
        #pragma unroll
        for (int e = 0; e < 8; ++e) fb.v[e] = (__bf16)wp0[e];
      }
      fb.u[4] = 0; fb.u[5] = 0; fb.u[6] = 0; fb.u[7] = 0;

      v8f acc;
      #pragma unroll
      for (int r = 0; r < 8; ++r) acc[r] = 0.0f;
      acc = __builtin_amdgcn_wmma_f32_16x16x32_bf16(
          false, fa.v, false, fb.v, (short)0, acc, false, false);

      // spill C tile; row br occupies 16 contiguous floats in scratch
      #pragma unroll
      for (int r = 0; r < 8; ++r) mysc[r * 32 + lane] = acc[r];
      // same-wave DS ops are in-order: loads below see the stores above
      const float* vrow = vs + ((size_t)br * 32 + kk) * 16 + hh * 8;
      float p = 0.0f, qq = 0.0f;
      #pragma unroll
      for (int d = 0; d < 8; ++d) {
        const float u = crow0[d];
        p  += u * u;
        qq += u * vrow[d];
      }
      p  += __shfl_xor(p, 16, 32);
      qq += __shfl_xor(qq, 16, 32);
      if (hh == 0) {                     // 16 lanes, one predicated store
        const float s2    = p;
        const float scale = sqrtf(s2) / (0.5f + s2);
        const float dist  = scale * scale * s2 - 2.0f * scale * qq
                            + vsq[br * 32 + kk];
        const size_t bi = (((size_t)(btile * 16 + br)) * NI + i) * KO + k;
        blog[bi] += 1.0f - dist;
      }
    }
  }
}

// ---------------------------------------------------------------------------
extern "C" void kernel_launch(void* const* d_in, const int* in_sizes, int n_in,
                              void* d_out, int out_size, void* d_ws, size_t ws_size,
                              hipStream_t stream) {
  const float* x = (const float*)d_in[0];
  // d_in[1] = y : unused; d_in[2] = labels = arange(K) -> W = W1 directly
  const float* W = (const float*)d_in[3];
  float* out = (float*)d_out;

  float* cbuf = (float*)d_ws;                    // BB*NI*KO
  float* blog = cbuf + (size_t)BB * NI * KO;     // BB*NI*KO
  float* vraw = blog + (size_t)BB * NI * KO;     // BB*KO*CO
  float* vbuf = vraw + (size_t)BB * KO * CO;     // BB*KO*CO
  float* Sbuf = vbuf + (size_t)BB * KO * CO;     // BB*KO
  const size_t baseFloats = (size_t)BB * NI * KO * 2 + (size_t)BB * KO * CO * 2 + BB * KO;
  uint32_t* Wswz = (uint32_t*)((char*)d_ws + baseFloats * 4);
  const size_t needBytes = baseFloats * 4 + (size_t)NI * KO * CO * CI * 2;
  const bool swz = (ws_size >= needBytes);

  const int BNK = BB * NI * KO;
  zero_kernel<<<(BNK + 255) / 256, 256, 0, stream>>>(blog, BNK);
  if (swz) swizzle_kernel<<<(NI / 2) * KO * 32 / 256, 256, 0, stream>>>(W, Wswz);

  #define VPASS(cb, cu) do {                                                   \
    if (swz) vpass_kernel<true><<<dim3(KO, 2), 256, 0, stream>>>(x, W, Wswz, cb, cu, vraw); \
    else     vpass_kernel<false><<<dim3(KO, 2), 256, 0, stream>>>(x, W, Wswz, cb, cu, vraw); \
  } while (0)
  #define DDPASS() do {                                                        \
    if (swz) ddpass_kernel<true><<<dim3(2, 4, 32), 256, 0, stream>>>(x, W, Wswz, vbuf, blog); \
    else     ddpass_kernel<false><<<dim3(2, 4, 32), 256, 0, stream>>>(x, W, Wswz, vbuf, blog); \
  } while (0)

  // iteration 0: normalized routing weights are uniform 1/NI
  VPASS((const float*)nullptr, 1.0f / NI);
  finalize_kernel<<<(BB * KO + 255) / 256, 256, 0, stream>>>(vraw, nullptr, vbuf, nullptr);
  DDPASS();

  for (int it = 1; it < 3; ++it) {
    zero_kernel<<<(BB * KO + 255) / 256, 256, 0, stream>>>(Sbuf, BB * KO);
    softmax_kernel<<<BB * NI, 128, 0, stream>>>(blog, cbuf, Sbuf);
    VPASS(cbuf, 0.0f);
    finalize_kernel<<<(BB * KO + 255) / 256, 256, 0, stream>>>(
        vraw, Sbuf, vbuf, (it == 2) ? out : nullptr);
    if (it == 1) DDPASS();
  }
  (void)in_sizes; (void)n_in; (void)out_size; (void)ws_size;
  #undef VPASS
  #undef DDPASS
}
